// MultiHeadAttention_68229850464681
// MI455X (gfx1250) — matrix operand order repaired and checked by reading, not compiled, not run on hardware
//
#include <hip/hip_runtime.h>

#define DEVFN __device__ __forceinline__

typedef _Float16 v16h __attribute__((ext_vector_type(16)));
typedef _Float16 v8h  __attribute__((ext_vector_type(8)));
typedef float    v8f  __attribute__((ext_vector_type(8)));
typedef float    v4f  __attribute__((ext_vector_type(4)));

constexpr int Bn  = 2;
constexpr int Ln  = 2048;
constexpr int Dn  = 1024;
constexpr int Hn  = 16;
constexpr int DKn = 64;
constexpr int BLn = Bn * Ln;        // 4096

// ---------------------------------------------------------------------------
// WMMA wrapper: D = A(16x32 f16) * B(32x16 f16) + C(16x16 f32)
// ---------------------------------------------------------------------------
DEVFN v8f wmma_f16(v16h a, v16h b, v8f c) {
  return __builtin_amdgcn_wmma_f32_16x16x32_f16(
      /*neg_a=*/false, a, /*neg_b=*/false, b,
      /*c_mod=*/(short)0, c, /*reuse_a=*/false, /*reuse_b=*/false);
}

DEVFN v8f zero8() {
  v8f z;
#pragma unroll
  for (int i = 0; i < 8; ++i) z[i] = 0.0f;
  return z;
}

// ---------------------------------------------------------------------------
// Fragment loaders (layouts per CDNA5 ISA §7.12.2, wave32)
// A 16x32 f16: lane<16 -> row=lane, elems[0..7]=K(kh..kh+7), [8..15]=K(kh+16..)
//              (caller passes p already offset by row*ld + k0 + kh)
// B 32x16 f16: lane<16 -> col=lane, elems[0..7]=K(koff..koff+7), [8..15]=K(koff+16..)
// ---------------------------------------------------------------------------
DEVFN v16h load_a_h(const _Float16* __restrict__ p) {
  v8h u0 = *(const v8h*)(p);
  v8h u1 = *(const v8h*)(p + 16);
  v16h a;
#pragma unroll
  for (int i = 0; i < 8; ++i) { a[i] = u0[i]; a[8 + i] = u1[i]; }
  return a;
}

DEVFN v16h load_b_h(const _Float16* __restrict__ p) {
  v8h u0 = *(const v8h*)(p);
  v8h u1 = *(const v8h*)(p + 16);
  v16h b;
#pragma unroll
  for (int i = 0; i < 8; ++i) { b[i] = u0[i]; b[8 + i] = u1[i]; }
  return b;
}

// ---------------------------------------------------------------------------
// Kernel 0: one-shot f32 -> f16 conversion (8 elements / lane, b128 in,
// b128 out).  Each element converted exactly once (vs 8-32x if converted
// inside the GEMM inner loops).
// ---------------------------------------------------------------------------
__global__ __launch_bounds__(256) void cvt_f16_kernel(
    const float* __restrict__ src, _Float16* __restrict__ dst) {
  const size_t i = ((size_t)blockIdx.x * blockDim.x + threadIdx.x) * 8;
  v4f a = *(const v4f*)(src + i);
  v4f b = *(const v4f*)(src + i + 4);
  v8h o;
#pragma unroll
  for (int k = 0; k < 4; ++k) {
    o[k]     = (_Float16)a[k];
    o[4 + k] = (_Float16)b[k];
  }
  *(v8h*)(dst + i) = o;
}

// ---------------------------------------------------------------------------
// Kernel 1: Q/K/V projections.  Y = Xh @ Wh^T + bias  (f16 in, f16 out,
// f32 accumulate).  blockIdx.z selects {q,k,v}.  Wave computes 64x64 tile.
// Q,K stored (B,H,L,DK) row-major; V stored transposed (B,H,DK,L).
// ---------------------------------------------------------------------------
__global__ __launch_bounds__(128) void qkv_proj_kernel(
    const _Float16* __restrict__ Xq, const _Float16* __restrict__ Xk,
    const _Float16* __restrict__ Xv,
    const _Float16* __restrict__ Wq, const float* __restrict__ bq,
    const _Float16* __restrict__ Wk, const float* __restrict__ bk,
    const _Float16* __restrict__ Wv, const float* __restrict__ bv_,
    _Float16* __restrict__ Qh, _Float16* __restrict__ Kh,
    _Float16* __restrict__ Vt) {
  const int mode = blockIdx.z;
  const _Float16* X; const _Float16* W; const float* bias; _Float16* out;
  if (mode == 0)      { X = Xq; W = Wq; bias = bq;  out = Qh; }
  else if (mode == 1) { X = Xk; W = Wk; bias = bk;  out = Kh; }
  else                { X = Xv; W = Wv; bias = bv_; out = Vt; }

  const int lane = threadIdx.x & 31;
  const int wave = threadIdx.x >> 5;
  const int m0 = blockIdx.x * 128 + (wave >> 1) * 64;   // rows in [0,4096)
  const int n0 = blockIdx.y * 128 + (wave & 1) * 64;    // cols in [0,1024)
  const int r    = lane & 15;
  const int kh   = (lane >> 4) << 3;   // A-frag K offset: 0 or 8
  const int koff = (lane >> 4) << 3;   // B-frag K offset: 0 or 8
  const int hi8  = kh;                 // C/D row offset:  0 or 8

  v8f acc[4][4];
#pragma unroll
  for (int i = 0; i < 4; ++i)
#pragma unroll
    for (int j = 0; j < 4; ++j) acc[i][j] = zero8();

  for (int kk = 0; kk < Dn; kk += 32) {
    v16h bf[4];
#pragma unroll
    for (int j = 0; j < 4; ++j)
      bf[j] = load_b_h(W + (size_t)(n0 + 16 * j + r) * Dn + kk + koff);
#pragma unroll
    for (int i = 0; i < 4; ++i) {
      v16h af = load_a_h(X + (size_t)(m0 + 16 * i + r) * Dn + kk + kh);
#pragma unroll
      for (int j = 0; j < 4; ++j) acc[i][j] = wmma_f16(af, bf[j], acc[i][j]);
    }
  }

#pragma unroll
  for (int j = 0; j < 4; ++j) {
    const int gn = n0 + 16 * j + r;
    const float bv = bias[gn];
    const int h  = gn >> 6;     // head
    const int dk = gn & 63;     // dim within head
#pragma unroll
    for (int i = 0; i < 4; ++i) {
#pragma unroll
      for (int v = 0; v < 8; ++v) {
        const int gm = m0 + 16 * i + v + hi8;
        const int bb = gm >> 11;         // batch
        const int l  = gm & 2047;        // sequence pos
        const float val = acc[i][j][v] + bv;
        size_t idx;
        if (mode < 2)
          idx = ((size_t)(bb * Hn + h) * Ln + l) * DKn + dk;       // (B,H,L,DK)
        else
          idx = ((size_t)(bb * Hn + h) * DKn + dk) * Ln + l;       // (B,H,DK,L)
        out[idx] = (_Float16)val;
      }
    }
  }
}

// ---------------------------------------------------------------------------
// Kernel 2: causal flash attention.  One wave per 16-query tile.
// S = (Q K^T)/8 -> online softmax -> O += P V.  P bounced via LDS to become
// an A-fragment.  grid = (L/64, B*H), block = 128 (4 waves).
// ---------------------------------------------------------------------------
__global__ __launch_bounds__(128) void flash_attn_kernel(
    const _Float16* __restrict__ Qh, const _Float16* __restrict__ Kh,
    const _Float16* __restrict__ Vt, _Float16* __restrict__ AO) {
  __shared__ _Float16 Pbuf[4][16][32];   // per-wave P tile (16 q x 32 k)

  const int lane = threadIdx.x & 31;
  const int wave = threadIdx.x >> 5;
  const int bh = blockIdx.y;
  const int b  = bh >> 4;
  const int h  = bh & 15;
  const int q0 = (blockIdx.x * 4 + wave) * 16;

  const _Float16* Qp = Qh + (size_t)bh * Ln * DKn;
  const _Float16* Kp = Kh + (size_t)bh * Ln * DKn;
  const _Float16* Vp = Vt + (size_t)bh * DKn * Ln;

  const int r    = lane & 15;
  const int kh   = (lane >> 4) << 3;
  const int koff = (lane >> 4) << 3;
  const int hi8  = kh;

  // Q fragments for dk 0..31 and 32..63 (stay resident whole loop)
  const v16h qf0 = load_a_h(Qp + (size_t)(q0 + r) * DKn + 0 + kh);
  const v16h qf1 = load_a_h(Qp + (size_t)(q0 + r) * DKn + 32 + kh);

  v8f o[4];
#pragma unroll
  for (int j = 0; j < 4; ++j) o[j] = zero8();
  float mrow[8], lrow[8];
#pragma unroll
  for (int v = 0; v < 8; ++v) { mrow[v] = -3.0e30f; lrow[v] = 0.0f; }

  const int nk = (q0 + 47) >> 5;   // number of 32-key tiles (causal)
  for (int it = 0; it < nk; ++it) {
    const int k0 = it << 5;

    // S tiles: two 16x16 f32 tiles covering keys [k0,k0+32)
    v8f s0 = zero8(), s1 = zero8();
    {
      v16h bkf = load_b_h(Kp + (size_t)(k0 + r) * DKn + koff);
      s0 = wmma_f16(qf0, bkf, s0);
      bkf = load_b_h(Kp + (size_t)(k0 + r) * DKn + 32 + koff);
      s0 = wmma_f16(qf1, bkf, s0);
      bkf = load_b_h(Kp + (size_t)(k0 + 16 + r) * DKn + koff);
      s1 = wmma_f16(qf0, bkf, s1);
      bkf = load_b_h(Kp + (size_t)(k0 + 16 + r) * DKn + 32 + koff);
      s1 = wmma_f16(qf1, bkf, s1);
    }

    // online softmax update, row by row (row = v + hi8)
#pragma unroll
    for (int v = 0; v < 8; ++v) {
      const int gq = q0 + v + hi8;
      float x0 = s0[v] * 0.125f;           // 1/sqrt(64)
      float x1 = s1[v] * 0.125f;
      if (k0 + r > gq)      x0 = -3.0e30f; // causal mask
      if (k0 + 16 + r > gq) x1 = -3.0e30f;
      float mx = fmaxf(x0, x1);
      mx = fmaxf(mx, __shfl_xor(mx, 1));
      mx = fmaxf(mx, __shfl_xor(mx, 2));
      mx = fmaxf(mx, __shfl_xor(mx, 4));
      mx = fmaxf(mx, __shfl_xor(mx, 8));
      const float mnew = fmaxf(mrow[v], mx);
      const float corr = __expf(mrow[v] - mnew);
      mrow[v] = mnew;
      const float p0 = __expf(x0 - mnew);
      const float p1 = __expf(x1 - mnew);
      float rs = p0 + p1;
      rs += __shfl_xor(rs, 1);
      rs += __shfl_xor(rs, 2);
      rs += __shfl_xor(rs, 4);
      rs += __shfl_xor(rs, 8);
      lrow[v] = lrow[v] * corr + rs;
#pragma unroll
      for (int j = 0; j < 4; ++j) o[j][v] *= corr;
      const int row = v + hi8;
      Pbuf[wave][row][r]      = (_Float16)p0;
      Pbuf[wave][row][r + 16] = (_Float16)p1;
    }

    // same-wave LDS RAW: wait for DS stores, block compiler reordering
    asm volatile("s_wait_dscnt 0x0" ::: "memory");

    // P as A-fragment (16x32), V^T rows as B-fragments -> O += P*V
    const v16h pa = load_a_h(&Pbuf[wave][r][kh]);
#pragma unroll
    for (int j = 0; j < 4; ++j) {
      v16h bvf = load_b_h(Vp + (size_t)(16 * j + r) * Ln + k0 + koff);
      o[j] = wmma_f16(pa, bvf, o[j]);
    }
  }

  // normalize and store to (B,L,D) f16
#pragma unroll
  for (int v = 0; v < 8; ++v) {
    const float inv = 1.0f / lrow[v];
    const int row = q0 + v + hi8;
#pragma unroll
    for (int j = 0; j < 4; ++j) {
      AO[((size_t)b * Ln + row) * Dn + h * DKn + 16 * j + r] =
          (_Float16)(o[j][v] * inv);
    }
  }
}

// ---------------------------------------------------------------------------
// Kernel 3: output projection.  out = AO @ Wo^T + b_o  (f16 in, f32 out)
// ---------------------------------------------------------------------------
__global__ __launch_bounds__(128) void out_proj_kernel(
    const _Float16* __restrict__ A, const _Float16* __restrict__ W,
    const float* __restrict__ bias, float* __restrict__ Y) {
  const int lane = threadIdx.x & 31;
  const int wave = threadIdx.x >> 5;
  const int m0 = blockIdx.x * 128 + (wave >> 1) * 64;
  const int n0 = blockIdx.y * 128 + (wave & 1) * 64;
  const int r    = lane & 15;
  const int kh   = (lane >> 4) << 3;
  const int koff = (lane >> 4) << 3;
  const int hi8  = kh;

  v8f acc[4][4];
#pragma unroll
  for (int i = 0; i < 4; ++i)
#pragma unroll
    for (int j = 0; j < 4; ++j) acc[i][j] = zero8();

  for (int kk = 0; kk < Dn; kk += 32) {
    v16h bf[4];
#pragma unroll
    for (int j = 0; j < 4; ++j)
      bf[j] = load_b_h(W + (size_t)(n0 + 16 * j + r) * Dn + kk + koff);
#pragma unroll
    for (int i = 0; i < 4; ++i) {
      v16h af = load_a_h(A + (size_t)(m0 + 16 * i + r) * Dn + kk + kh);
#pragma unroll
      for (int j = 0; j < 4; ++j) acc[i][j] = wmma_f16(af, bf[j], acc[i][j]);
    }
  }

#pragma unroll
  for (int j = 0; j < 4; ++j) {
    const int gn = n0 + 16 * j + r;
    const float bv = bias[gn];
#pragma unroll
    for (int i = 0; i < 4; ++i) {
#pragma unroll
      for (int v = 0; v < 8; ++v) {
        const int gm = m0 + 16 * i + v + hi8;
        Y[(size_t)gm * Dn + gn] = acc[i][j][v] + bv;
      }
    }
  }
}

// ---------------------------------------------------------------------------
// Host launch
// ---------------------------------------------------------------------------
extern "C" void kernel_launch(void* const* d_in, const int* in_sizes, int n_in,
                              void* d_out, int out_size, void* d_ws,
                              size_t ws_size, hipStream_t stream) {
  const float* query = (const float*)d_in[0];
  const float* key_  = (const float*)d_in[1];
  const float* value = (const float*)d_in[2];
  // d_in[3] = causal mask (bool) -- implemented analytically in-kernel
  const float* w_q = (const float*)d_in[4];
  const float* b_q = (const float*)d_in[5];
  const float* w_k = (const float*)d_in[6];
  const float* b_k = (const float*)d_in[7];
  const float* w_v = (const float*)d_in[8];
  const float* b_v = (const float*)d_in[9];
  const float* w_o = (const float*)d_in[10];
  const float* b_o = (const float*)d_in[11];

  const size_t act  = (size_t)BLn * Dn;          // 4M elements (8 MB f16)
  const size_t wgt  = (size_t)Dn * Dn;           // 1M elements (2 MB f16)
  char* ws = (char*)d_ws;
  _Float16* Qh = (_Float16*)(ws + 0 * act * 2);
  _Float16* Kh = (_Float16*)(ws + 1 * act * 2);
  _Float16* Vt = (_Float16*)(ws + 2 * act * 2);
  _Float16* AO = (_Float16*)(ws + 3 * act * 2);
  _Float16* Xq = (_Float16*)(ws + 4 * act * 2);
  _Float16* Xk = (_Float16*)(ws + 5 * act * 2);
  _Float16* Xv = (_Float16*)(ws + 6 * act * 2);
  _Float16* Wq = (_Float16*)(ws + 7 * act * 2 + 0 * wgt * 2);
  _Float16* Wk = (_Float16*)(ws + 7 * act * 2 + 1 * wgt * 2);
  _Float16* Wv = (_Float16*)(ws + 7 * act * 2 + 2 * wgt * 2);
  _Float16* Wo = (_Float16*)(ws + 7 * act * 2 + 3 * wgt * 2);

  // one-shot f32 -> f16 conversions (each element exactly once)
  const int actBlocks = (int)(act / (8 * 256));  // 2048
  const int wgtBlocks = (int)(wgt / (8 * 256));  // 512
  cvt_f16_kernel<<<actBlocks, 256, 0, stream>>>(query, Xq);
  cvt_f16_kernel<<<actBlocks, 256, 0, stream>>>(key_,  Xk);
  cvt_f16_kernel<<<actBlocks, 256, 0, stream>>>(value, Xv);
  cvt_f16_kernel<<<wgtBlocks, 256, 0, stream>>>(w_q, Wq);
  cvt_f16_kernel<<<wgtBlocks, 256, 0, stream>>>(w_k, Wk);
  cvt_f16_kernel<<<wgtBlocks, 256, 0, stream>>>(w_v, Wv);
  cvt_f16_kernel<<<wgtBlocks, 256, 0, stream>>>(w_o, Wo);

  qkv_proj_kernel<<<dim3(BLn / 128, Dn / 128, 3), 128, 0, stream>>>(
      Xq, Xk, Xv, Wq, b_q, Wk, b_k, Wv, b_v, Qh, Kh, Vt);

  flash_attn_kernel<<<dim3(Ln / 64, Bn * Hn), 128, 0, stream>>>(Qh, Kh, Vt, AO);

  out_proj_kernel<<<dim3(BLn / 128, Dn / 128), 128, 0, stream>>>(
      AO, Wo, b_o, (float*)d_out);
}
